// BiGRU_4389456576647
// MI455X (gfx1250) — compile-verified, matrix operand order checked by reading, not run on hardware
//
#include <hip/hip_runtime.h>
#include <hip/hip_bf16.h>

// ---------------------------------------------------------------------------
// BiGRU (Keras reset_after=True) on gfx1250 using v_wmma_f32_16x16x32_bf16.
// B=64, T=512, E=300, U=256, G=768, C=20, V=50000.
// ---------------------------------------------------------------------------

typedef __attribute__((ext_vector_type(8)))  float          v8f;
typedef __attribute__((ext_vector_type(8)))  unsigned short v8us;
typedef __attribute__((ext_vector_type(16))) unsigned short v16us;
typedef __attribute__((ext_vector_type(16))) __bf16         v16bf;

#define WMMA_BF16(a, b, c) \
    __builtin_amdgcn_wmma_f32_16x16x32_bf16(false, (a), false, (b), (short)0, (c), false, false)

static __device__ __forceinline__ unsigned short f2bf_us(float v) {
    union { float f; unsigned u; } cv; cv.f = v;
    unsigned r = cv.u + 0x7FFFu + ((cv.u >> 16) & 1u);   // RNE
    return (unsigned short)(r >> 16);
}
static __device__ __forceinline__ float bf2f(unsigned short h) {
    union { unsigned u; float f; } cv; cv.u = ((unsigned)h) << 16;
    return cv.f;
}

// Fragment layout helper (ISA 7.12.2, 16-bit A/B 16x32):
//   lane<16 : j<8 -> K=k0+j,       j>=8 -> K=k0+16+(j-8)
//   lane>=16: j<8 -> K=k0+8+j,     j>=8 -> K=k0+24+(j-8)
static __device__ __forceinline__ int frag_k(int k0, int hs, int j) {
    return k0 + hs * 8 + ((j < 8) ? j : (16 + j - 8));
}

// ---------------------------------------------------------------------------
// Convert f32 src[R x Kact] -> bf16 dst[R x Kpad], zero-padding columns.
// ---------------------------------------------------------------------------
__global__ void cvt_pad_bf16_kernel(const float* __restrict__ src, unsigned short* __restrict__ dst,
                                    int Kact, int Kpad, long total /* R*Kpad */) {
    long tid = (long)blockIdx.x * blockDim.x + threadIdx.x;
    if (tid >= total) return;
    long row = tid / Kpad;
    int  col = (int)(tid - row * Kpad);
    float v = (col < Kact) ? src[row * Kact + col] : 0.0f;
    dst[tid] = f2bf_us(v);
}

// ---------------------------------------------------------------------------
// Pre-swizzle a weight matrix w[Kact x N] (f32, row-major) into the per-lane
// bf16 WMMA B-fragment layout, zero-padding K to KT*32.
// dst index: ((k_tile*NT + n_tile)*32 + lane)*16 + j  (32B contiguous / lane)
// ---------------------------------------------------------------------------
__global__ void swz_b_kernel(const float* __restrict__ w, unsigned short* __restrict__ dst,
                             int Kact, int KT, int N) {
    int tid = blockIdx.x * blockDim.x + threadIdx.x;
    int NT = N / 16;
    if (tid >= KT * NT * 32) return;
    int lane = tid & 31, tile = tid >> 5;
    int n_tile = tile % NT, k_tile = tile / NT;
    int ln = lane & 15, hs = lane >> 4;
    int n = n_tile * 16 + ln;
    int k0 = k_tile * 32;
#pragma unroll
    for (int j = 0; j < 16; ++j) {
        int k = frag_k(k0, hs, j);
        float v = (k < Kact) ? w[(long)k * N + n] : 0.0f;
        dst[(long)tid * 16 + j] = f2bf_us(v);
    }
}

// ---------------------------------------------------------------------------
// xw GEMM: out[M x 768] = A_bf16 @ Bsw + bias.  M = B*T = 32768.
// USE_IDX=1: A row for output row m is a_bf[x[m]] (padded bf16 emb table)
// USE_IDX=0: A row is a_bf[m] (h1 buffer). One wave per 16x16 tile.
// ---------------------------------------------------------------------------
template <int USE_IDX>
__global__ void xw_gemm_kernel(const int* __restrict__ x,
                               const unsigned short* __restrict__ a_bf, int a_stride,
                               const unsigned short* __restrict__ b_sw,
                               const float* __restrict__ bias, float* __restrict__ out,
                               int Mtiles, int KT) {
    const int NT = 48;                       // 768 / 16
    int wave = threadIdx.x >> 5, lane = threadIdx.x & 31;
    int lin = blockIdx.x * 8 + wave;         // wave-uniform -> EXEC all-1 in WMMA
    if (lin >= Mtiles * NT) return;
    int n_tile = lin % NT, m_tile = lin / NT;
    int ln = lane & 15, hs = lane >> 4;
    int m = m_tile * 16 + ln;                // output row this lane serves

    int row = USE_IDX ? x[m] : m;
    const unsigned short* arow = a_bf + (long)row * a_stride;
    const unsigned short* bptr = b_sw + ((long)n_tile * 32 + lane) * 16;

    v8f acc = {0.f, 0.f, 0.f, 0.f, 0.f, 0.f, 0.f, 0.f};
    for (int kt = 0; kt < KT; ++kt) {
        int klo = kt * 32 + hs * 8;
        v8us lo = *(const v8us*)(arow + klo);
        v8us hi = *(const v8us*)(arow + klo + 16);
        v16us a_us;
#pragma unroll
        for (int i = 0; i < 8; ++i) { a_us[i] = lo[i]; a_us[i + 8] = hi[i]; }
        v16us b_us = *(const v16us*)(bptr + (long)kt * NT * 32 * 16);
        acc = WMMA_BF16(__builtin_bit_cast(v16bf, a_us), __builtin_bit_cast(v16bf, b_us), acc);
    }
    int col = n_tile * 16 + ln;
    float bv = bias[col];
    int rowb = m_tile * 16 + hs * 8;         // C/D layout: VGPR i -> M = rowb + i
#pragma unroll
    for (int i = 0; i < 8; ++i)
        out[(long)(rowb + i) * 768 + col] = acc[i] + bv;
}

// ---------------------------------------------------------------------------
// Recurrent GRU scan: one persistent 1024-thread workgroup (32 waves) over T.
// Per step: inner[64,768] = h[64,256] @ rk + b[1]  (192 tiles = 32 waves x 6),
// then elementwise gates, h update in LDS, barrier.
// kt loop kept rolled + B pointer laundered through asm so the L2-hot weight
// fragments are re-loaded per step (no hoist -> no scratch spill).
// ---------------------------------------------------------------------------
__global__ void gru_scan_kernel(const float* __restrict__ xw,          // [B*T, 768] (+b[0] folded)
                                const unsigned short* __restrict__ rk_sw,
                                const float* __restrict__ rbias,       // b[1], len 768
                                unsigned short* __restrict__ out_all,  // bf16 [B*T, out_stride]
                                int out_stride, int out_col,
                                float* __restrict__ out_final,         // f32 [B, 512]
                                int final_col,
                                int T, int reverse, int store_all, int store_final) {
    const int Bn = 64, U = 256, G = 768, NT = 48;
    const int NTHREADS = 1024;
    __shared__ unsigned short h_lds[Bn * U];        // 32 KB bf16 hidden state
    extern __shared__ float inner[];                // 192 KB dynamic: Bn*G f32

    int tid = threadIdx.x, wave = tid >> 5, lane = tid & 31;
    int ln = lane & 15, hs = lane >> 4;
    int m_tile = wave & 3, n_grp = wave >> 2;       // 4 M-tiles x 8 N-groups(x6)

    for (int e = tid; e < Bn * U; e += NTHREADS) h_lds[e] = 0;
    __syncthreads();

    const unsigned short* bbase = rk_sw + (((long)n_grp * 6) * 32 + lane) * 16;

    for (int step = 0; step < T; ++step) {
        int t = reverse ? (T - 1 - step) : step;

        v8f acc[6];
#pragma unroll
        for (int j = 0; j < 6; ++j) acc[j] = (v8f){0.f, 0.f, 0.f, 0.f, 0.f, 0.f, 0.f, 0.f};

#pragma unroll 1
        for (int kt = 0; kt < 8; ++kt) {
            int m = m_tile * 16 + ln;
            int klo = kt * 32 + hs * 8;
            v8us lo = *(const v8us*)&h_lds[m * U + klo];
            v8us hi = *(const v8us*)&h_lds[m * U + klo + 16];
            v16us a_us;
#pragma unroll
            for (int i = 0; i < 8; ++i) { a_us[i] = lo[i]; a_us[i + 8] = hi[i]; }
            v16bf a = __builtin_bit_cast(v16bf, a_us);

            // opaque pointer: force per-step reload from global (L2-hot),
            // preventing loop-invariant hoist + scratch spill
            const unsigned short* bk = bbase + (long)kt * NT * 32 * 16;
            asm volatile("" : "+v"(bk));

            v16us bfr[6];
#pragma unroll
            for (int j = 0; j < 6; ++j)
                bfr[j] = *(const v16us*)(bk + (long)j * 32 * 16);
#pragma unroll
            for (int j = 0; j < 6; ++j)
                acc[j] = WMMA_BF16(a, __builtin_bit_cast(v16bf, bfr[j]), acc[j]);
        }
        // scatter inner (+ recurrent bias) to LDS
#pragma unroll
        for (int j = 0; j < 6; ++j) {
            int n_tile = n_grp * 6 + j;
            int col = n_tile * 16 + ln;
            int rowb = m_tile * 16 + hs * 8;
            float bv = rbias[col];
#pragma unroll
            for (int i = 0; i < 8; ++i)
                inner[(rowb + i) * G + col] = acc[j][i] + bv;
        }
        __syncthreads();

        // gates: 64*256 elements / 1024 threads = 16 each
        for (int r = 0; r < 16; ++r) {
            int e = r * NTHREADS + tid;
            int b = e >> 8, u = e & 255;
            const float* xp = xw + (long)(b * T + t) * G;
            float gz = xp[u]       + inner[b * G + u];
            float gr = xp[256 + u] + inner[b * G + 256 + u];
            float xh = xp[512 + u];
            float rh = inner[b * G + 512 + u];
            float z  = 1.f / (1.f + __expf(-gz));
            float rr = 1.f / (1.f + __expf(-gr));
            float hh = tanhf(xh + rr * rh);
            float hp = bf2f(h_lds[b * U + u]);
            float hn = z * hp + (1.f - z) * hh;
            h_lds[b * U + u] = f2bf_us(hn);
            if (store_all)
                out_all[(long)(b * T + t) * out_stride + out_col + u] = f2bf_us(hn);
        }
        __syncthreads();
    }

    if (store_final) {
        for (int r = 0; r < 16; ++r) {
            int e = r * NTHREADS + tid;
            int b = e >> 8, u = e & 255;
            out_final[b * 512 + final_col + u] = bf2f(h_lds[b * U + u]);
        }
    }
}

// ---------------------------------------------------------------------------
// logits + softmax: out[64,20] = softmax(h2[64,512] @ wout[512,20] + bout)
// ---------------------------------------------------------------------------
__global__ void out_kernel(const float* __restrict__ h2, const float* __restrict__ wout,
                           const float* __restrict__ bout, float* __restrict__ out) {
    int b = blockIdx.x, c = threadIdx.x;
    __shared__ float lg[32], ex[32];
    if (c < 20) {
        float acc = bout[c];
        for (int k = 0; k < 512; ++k) acc += h2[b * 512 + k] * wout[k * 20 + c];
        lg[c] = acc;
    }
    __syncthreads();
    if (c < 20) {
        float mx = -1e30f;
        for (int i = 0; i < 20; ++i) mx = fmaxf(mx, lg[i]);
        ex[c] = __expf(lg[c] - mx);
    }
    __syncthreads();
    if (c < 20) {
        float s = 0.f;
        for (int i = 0; i < 20; ++i) s += ex[i];
        out[b * 20 + c] = ex[c] / s;
    }
}

// ---------------------------------------------------------------------------
extern "C" void kernel_launch(void* const* d_in, const int* in_sizes, int n_in,
                              void* d_out, int out_size, void* d_ws, size_t ws_size,
                              hipStream_t stream) {
    (void)in_sizes; (void)n_in; (void)out_size; (void)ws_size;
    const int B = 64, T = 512, G = 768, V = 50000;
    const long BT = (long)B * T;

    const int*   x    = (const int*)  d_in[0];
    const float* emb  = (const float*)d_in[1];
    const float* k1f  = (const float*)d_in[2];
    const float* rk1f = (const float*)d_in[3];
    const float* b1f  = (const float*)d_in[4];
    const float* k1b  = (const float*)d_in[5];
    const float* rk1b = (const float*)d_in[6];
    const float* b1b  = (const float*)d_in[7];
    const float* k2f  = (const float*)d_in[8];
    const float* rk2f = (const float*)d_in[9];
    const float* b2f  = (const float*)d_in[10];
    const float* k2b  = (const float*)d_in[11];
    const float* rk2b = (const float*)d_in[12];
    const float* b2b  = (const float*)d_in[13];
    const float* wout = (const float*)d_in[14];
    const float* bout = (const float*)d_in[15];

    // workspace carving (256B aligned)
    char* ws = (char*)d_ws;
    size_t off = 0;
    auto carve = [&](size_t bytes) -> char* {
        char* p = ws + off;
        off = (off + bytes + 255) & ~(size_t)255;
        return p;
    };
    const size_t tileB = 32 * 16 * 2;   // bytes per swizzled fragment tile
    unsigned short* k1f_sw  = (unsigned short*)carve(10 * 48 * tileB);
    unsigned short* k1b_sw  = (unsigned short*)carve(10 * 48 * tileB);
    unsigned short* rk1f_sw = (unsigned short*)carve( 8 * 48 * tileB);
    unsigned short* rk1b_sw = (unsigned short*)carve( 8 * 48 * tileB);
    unsigned short* k2f_sw  = (unsigned short*)carve(16 * 48 * tileB);
    unsigned short* k2b_sw  = (unsigned short*)carve(16 * 48 * tileB);
    unsigned short* rk2f_sw = (unsigned short*)carve( 8 * 48 * tileB);
    unsigned short* rk2b_sw = (unsigned short*)carve( 8 * 48 * tileB);
    unsigned short* emb_bf  = (unsigned short*)carve((size_t)V * 320 * 2);
    float* xw1f = (float*)carve(BT * G * 4);
    float* xw1b = (float*)carve(BT * G * 4);
    float* xw2f = (float*)carve(BT * G * 4);
    float* xw2b = (float*)carve(BT * G * 4);
    unsigned short* h1 = (unsigned short*)carve(BT * 512 * 2);
    float* h2 = (float*)carve(B * 512 * 4);

    // 0) embedding table -> padded bf16 [V, 320]
    {
        long total = (long)V * 320;
        cvt_pad_bf16_kernel<<<(int)((total + 255) / 256), 256, 0, stream>>>(emb, emb_bf, 300, 320, total);
    }

    // 1) pre-swizzle weights to WMMA fragment layout (bf16)
    auto swz = [&](const float* w, unsigned short* dst, int Kact, int KT) {
        int total = KT * 48 * 32;
        swz_b_kernel<<<(total + 255) / 256, 256, 0, stream>>>(w, dst, Kact, KT, G);
    };
    swz(k1f,  k1f_sw,  300, 10);  swz(k1b,  k1b_sw,  300, 10);
    swz(rk1f, rk1f_sw, 256,  8);  swz(rk1b, rk1b_sw, 256,  8);
    swz(k2f,  k2f_sw,  512, 16);  swz(k2b,  k2b_sw,  512, 16);
    swz(rk2f, rk2f_sw, 256,  8);  swz(rk2b, rk2b_sw, 256,  8);

    // 2) layer-1 input projections (embedding row indirection in A fragments)
    const int Mtiles = (int)(BT / 16);                 // 2048
    const int gemmBlocks = Mtiles * 48 / 8;            // 12288
    xw_gemm_kernel<1><<<gemmBlocks, 256, 0, stream>>>(x, emb_bf, 320, k1f_sw, b1f, xw1f, Mtiles, 10);
    xw_gemm_kernel<1><<<gemmBlocks, 256, 0, stream>>>(x, emb_bf, 320, k1b_sw, b1b, xw1b, Mtiles, 10);

    // 3) layer-1 scans -> h1 (bf16, [B,T,512]; fwd cols 0..255, bwd cols 256..511)
    const size_t innerBytes = 64 * 768 * 4;
    gru_scan_kernel<<<1, 1024, innerBytes, stream>>>(xw1f, rk1f_sw, b1f + G, h1, 512, 0,
                                                     h2, 0, T, 0, 1, 0);
    gru_scan_kernel<<<1, 1024, innerBytes, stream>>>(xw1b, rk1b_sw, b1b + G, h1, 512, 256,
                                                     h2, 0, T, 1, 1, 0);

    // 4) layer-2 input projections from h1 (bf16 A)
    xw_gemm_kernel<0><<<gemmBlocks, 256, 0, stream>>>(nullptr, h1, 512, k2f_sw, b2f, xw2f, Mtiles, 16);
    xw_gemm_kernel<0><<<gemmBlocks, 256, 0, stream>>>(nullptr, h1, 512, k2b_sw, b2b, xw2b, Mtiles, 16);

    // 5) layer-2 scans -> only final hidden states into h2[64,512]
    gru_scan_kernel<<<1, 1024, innerBytes, stream>>>(xw2f, rk2f_sw, b2f + G, h1, 512, 0,
                                                     h2, 0, T, 0, 0, 1);
    gru_scan_kernel<<<1, 1024, innerBytes, stream>>>(xw2b, rk2b_sw, b2b + G, h1, 512, 0,
                                                     h2, 256, T, 1, 0, 1);

    // 6) classifier + softmax
    out_kernel<<<B, 32, 0, stream>>>(h2, wout, bout, (float*)d_out);
}
